// Attention_13426067767620
// MI455X (gfx1250) — matrix operand order repaired and checked by reading, not compiled, not run on hardware
//
#include <hip/hip_runtime.h>
#include <cstdint>

// ---------------------------------------------------------------------------
// Problem constants (fixed by setup_inputs)
// ---------------------------------------------------------------------------
#define B_    8
#define S_    1024
#define DIM_  512
#define H_    8
#define DH_   64
#define E_    1536          // 3*H*DH
#define NB_   33
#define NB2_  1089          // NB*NB
#define NB2P_ 1104          // padded to multiple of 16
#define QT_   32            // q rows per attention block

typedef __bf16 bf16_t;
typedef bf16_t        v16bf __attribute__((ext_vector_type(16)));
typedef float         v8f   __attribute__((ext_vector_type(8)));
typedef unsigned int  v4u   __attribute__((ext_vector_type(4)));
typedef int           v4i   __attribute__((ext_vector_type(4)));
typedef int           v8i   __attribute__((ext_vector_type(8)));

union Frag16 {
  v16bf          v;
  unsigned short h[16];
  v4u            q[2];
};

__device__ __forceinline__ unsigned short f2bf(float f) {
  union { float f; unsigned u; } x; x.f = f;
  unsigned u = x.u;
  u += 0x7FFFu + ((u >> 16) & 1u);   // round-to-nearest-even
  return (unsigned short)(u >> 16);
}

__device__ __forceinline__ v8f wmma_bf16(v16bf a, v16bf b, v8f c) {
  // v_wmma_f32_16x16x32_bf16  (neg_a, A, neg_b, B, c_mod, C, reuse_a, reuse_b)
  return __builtin_amdgcn_wmma_f32_16x16x32_bf16(false, a, false, b, (short)0, c, false, false);
}

// A operand (16x32 bf16). Source row-major [m][k], ld in halfs.
__device__ __forceinline__ v16bf load_a_bf(const unsigned short* base, int ld, int m0, int k0, int lane) {
  Frag16 f;
  int m  = m0 + (lane & 15);
  int kb = k0 + ((lane >> 4) << 3);
  const unsigned short* p = base + (size_t)m * ld + kb;
  f.q[0] = *(const v4u*)(p);        // K offsets kb..kb+7
  f.q[1] = *(const v4u*)(p + 16);   // K offsets kb+16..kb+23
  return f.v;
}

// A operand built from an f32 source (LDS attn probabilities) with on-the-fly bf16 convert.
__device__ __forceinline__ v16bf load_a_f32(const float* base, int ld, int m0, int k0, int lane) {
  Frag16 f;
  int m  = m0 + (lane & 15);
  int kb = k0 + ((lane >> 4) << 3);
  const float* p = base + (size_t)m * ld + kb;
  #pragma unroll
  for (int i = 0; i < 8; ++i) f.h[i]     = f2bf(p[i]);
  #pragma unroll
  for (int i = 0; i < 8; ++i) f.h[8 + i] = f2bf(p[16 + i]);
  return f.v;
}

// B operand (32x16 bf16). Source stored [n][k] row-major (k contiguous), ld in halfs.
__device__ __forceinline__ v16bf load_b_bf(const unsigned short* base, int ld, int n0, int k0, int lane) {
  Frag16 f;
  int n  = n0 + (lane & 15);
  int kb = k0 + ((lane >> 4) << 3);
  const unsigned short* p = base + (size_t)n * ld + kb;
  f.q[0] = *(const v4u*)(p);
  f.q[1] = *(const v4u*)(p + 16);
  return f.v;
}

// ---------------------------------------------------------------------------
// TDM: DMA a QT_ x DH_ bf16 tile (contiguous rows) from global to LDS.
// D# per CDNA5 ISA: group0 = {count/type/addrs}, group1 = dims/strides.
// ---------------------------------------------------------------------------
#if __has_builtin(__builtin_amdgcn_tensor_load_to_lds)
#define HAVE_TDM 1
__device__ __forceinline__ void tdm_load_q_tile(unsigned lds_addr, const unsigned short* gsrc) {
  unsigned long long ga = (unsigned long long)(uintptr_t)gsrc;
  v4u g0;
  g0[0] = 1u;                                                // count=1, user descriptor
  g0[1] = lds_addr;                                          // lds_addr [63:32]
  g0[2] = (unsigned)(ga & 0xFFFFFFFFu);                      // global_addr [95:64]
  g0[3] = (unsigned)((ga >> 32) & 0x01FFFFFFu) | (2u << 30); // global_addr hi, type=2
  v8i g1;
  g1[0] = (1 << 16);                       // data_size = 1 -> 2-byte elements
  g1[1] = (DH_ & 0xFFFF) << 16;            // tensor_dim0[15:0]   (bits 79:48 = 64)
  g1[2] = (QT_ & 0xFFFF) << 16;            // tensor_dim1[15:0]   (bits 111:80 = 32)
  g1[3] = (DH_ & 0xFFFF) << 16;            // tile_dim0           (bits 127:112 = 64)
  g1[4] = QT_;                             // tile_dim1 = 32, tile_dim2 = 0
  g1[5] = DH_;                             // tensor_dim0_stride = 64 (contiguous rows)
  g1[6] = 0; g1[7] = 0;
  v4i z4 = {0, 0, 0, 0};
#if defined(__clang_major__) && __clang_major__ >= 23
  v8i z8 = {0, 0, 0, 0, 0, 0, 0, 0};
  __builtin_amdgcn_tensor_load_to_lds(g0, g1, z4, z4, z8, 0);
#else
  __builtin_amdgcn_tensor_load_to_lds(g0, g1, z4, z4, 0);
#endif
}
#else
#define HAVE_TDM 0
#endif

// ---------------------------------------------------------------------------
// K0: f32 -> bf16 conversions (hidden, Wqkv, Wo, bias_table->padded)
// ---------------------------------------------------------------------------
__global__ void k_convert(const float* __restrict__ hs, const float* __restrict__ wqkv,
                          const float* __restrict__ wo, const float* __restrict__ bt,
                          unsigned short* __restrict__ hbf, unsigned short* __restrict__ wqkvbf,
                          unsigned short* __restrict__ wobf, unsigned short* __restrict__ btbf) {
  int stride = gridDim.x * blockDim.x;
  int tid = blockIdx.x * blockDim.x + threadIdx.x;
  for (int i = tid; i < B_ * S_ * DIM_; i += stride) hbf[i] = f2bf(hs[i]);
  for (int i = tid; i < E_ * DIM_;     i += stride) wqkvbf[i] = f2bf(wqkv[i]);
  for (int i = tid; i < DIM_ * DIM_;   i += stride) wobf[i] = f2bf(wo[i]);
  for (int i = tid; i < NB2P_ * DH_;   i += stride) {
    int j = i / DH_;
    btbf[i] = (j < NB2_) ? f2bf(bt[i]) : (unsigned short)0;  // zero pad rows
  }
}

// ---------------------------------------------------------------------------
// K1: QKV projection GEMM (8192 x 1536 x 512). Each wave: 16x64 output strip,
// software-pipelined fragments (next K-step preloaded into distinct registers
// so the b128 loads overlap the 4 WMMAs instead of serializing on one buffer).
// ---------------------------------------------------------------------------
__device__ __forceinline__ void scatter_qkv(v8f acc, int m0, int e_base, int lane,
                                            unsigned short* __restrict__ Qbf,
                                            unsigned short* __restrict__ Kbf,
                                            unsigned short* __restrict__ Vt) {
  int e = e_base + (lane & 15);
  int c = e % 3, hd = e / 3;
  int h = hd >> 6, d = hd & 63;
  int mhi = (lane >> 4) << 3;
  #pragma unroll
  for (int r = 0; r < 8; ++r) {
    int row = m0 + r + mhi;
    int b_ = row >> 10, s = row & (S_ - 1);
    unsigned short val = f2bf(acc[r]);
    size_t bh = (size_t)(b_ * H_ + h);
    if (c == 0)      Qbf[(bh * S_ + s) * DH_ + d] = val;
    else if (c == 1) Kbf[(bh * S_ + s) * DH_ + d] = val;
    else             Vt [(bh * DH_ + d) * S_ + s] = val;
  }
}

__global__ void k_qkv(const unsigned short* __restrict__ Hbf, const unsigned short* __restrict__ Wbf,
                      unsigned short* __restrict__ Qbf, unsigned short* __restrict__ Kbf,
                      unsigned short* __restrict__ Vt) {
  int wave = blockIdx.x * (blockDim.x >> 5) + (threadIdx.x >> 5);
  int lane = threadIdx.x & 31;
  const int tilesN = E_ / 64;   // 24 strips of 64 columns
  int mt = wave / tilesN, nt = wave - mt * tilesN;
  int m0 = mt * 16, n0 = nt * 64;
  v8f acc0 = {}, acc1 = {}, acc2 = {}, acc3 = {};
  v16bf a  = load_a_bf(Hbf, DIM_, m0, 0, lane);
  v16bf b0 = load_b_bf(Wbf, DIM_, n0,      0, lane);
  v16bf b1 = load_b_bf(Wbf, DIM_, n0 + 16, 0, lane);
  v16bf b2 = load_b_bf(Wbf, DIM_, n0 + 32, 0, lane);
  v16bf b3 = load_b_bf(Wbf, DIM_, n0 + 48, 0, lane);
  for (int k0 = 0; k0 < DIM_; k0 += 32) {
    int kn = (k0 + 32 < DIM_) ? (k0 + 32) : k0;   // branchless tail (redundant reload)
    v16bf an  = load_a_bf(Hbf, DIM_, m0, kn, lane);
    v16bf bn0 = load_b_bf(Wbf, DIM_, n0,      kn, lane);
    v16bf bn1 = load_b_bf(Wbf, DIM_, n0 + 16, kn, lane);
    v16bf bn2 = load_b_bf(Wbf, DIM_, n0 + 32, kn, lane);
    v16bf bn3 = load_b_bf(Wbf, DIM_, n0 + 48, kn, lane);
    __builtin_prefetch(Hbf + (size_t)(m0 + (lane & 15)) * DIM_ + k0 + 64, 0, 1);
    acc0 = wmma_bf16(a, b0, acc0);
    acc1 = wmma_bf16(a, b1, acc1);
    acc2 = wmma_bf16(a, b2, acc2);
    acc3 = wmma_bf16(a, b3, acc3);
    a = an; b0 = bn0; b1 = bn1; b2 = bn2; b3 = bn3;
  }
  scatter_qkv(acc0, m0, n0,      lane, Qbf, Kbf, Vt);
  scatter_qkv(acc1, m0, n0 + 16, lane, Qbf, Kbf, Vt);
  scatter_qkv(acc2, m0, n0 + 32, lane, Qbf, Kbf, Vt);
  scatter_qkv(acc3, m0, n0 + 48, lane, Qbf, Kbf, Vt);
}

// ---------------------------------------------------------------------------
// K2: fused attention. One block = (b, h, 32 q-rows). 256 threads = 8 waves.
// LDS: sc[32][1024] f32 (128KB), qb[32][1104] f32 (141KB), qt[32][64] bf16,
// rowsum[32], mk[32][1024] u8 mask tile (32KB).  Total ~309KB (<320KB/WGP).
// Q tile arrives via TDM (tensorcnt); mask tile via async b128 (asynccnt),
// overlapped with the qb/score WMMA phases.
// ---------------------------------------------------------------------------
__global__ void k_attn(const unsigned short* __restrict__ Qbf, const unsigned short* __restrict__ Kbf,
                       const unsigned short* __restrict__ Vt, const unsigned short* __restrict__ BTbf,
                       const unsigned char* __restrict__ mask, unsigned short* __restrict__ AObf,
                       const int* __restrict__ qs1p, const int* __restrict__ ks1p,
                       const int* __restrict__ topkp) {
  extern __shared__ char smem[];
  float* sc = (float*)smem;                                   // QT_*S_
  float* qb = sc + QT_ * S_;                                  // QT_*NB2P_
  unsigned short* qt = (unsigned short*)(qb + QT_ * NB2P_);   // QT_*DH_
  float* rowsum = (float*)(qt + QT_ * DH_);                   // QT_
  unsigned char* mk = (unsigned char*)(rowsum + QT_);         // QT_*S_ bytes

  int blk = blockIdx.x;
  int qtile = blk & 31;
  int bh = blk >> 5;
  int h = bh & 7, b = bh >> 3;
  int qbase = qtile * QT_;
  int tid = threadIdx.x, wave = tid >> 5, lane = tid & 31;
  int qs1 = *qs1p, ks1 = *ks1p, topk = *topkp;

  const unsigned short* Qrows = Qbf + ((size_t)bh * S_ + qbase) * DH_;
  const unsigned short* Krows = Kbf + (size_t)bh * S_ * DH_;
  const unsigned short* Vrows = Vt  + (size_t)bh * DH_ * S_;

  // phase 0a: Q tile -> LDS via Tensor Data Mover (one descriptor, wave 0).
#if HAVE_TDM
  if (wave == 0) {
    tdm_load_q_tile((unsigned)(uintptr_t)qt, Qrows);
  }
#else
  {
    int i = tid * 8;
    unsigned ldsa = (unsigned)(uintptr_t)(qt + i);
    unsigned long long ga = (unsigned long long)(uintptr_t)(Qrows + i);
    asm volatile("global_load_async_to_lds_b128 %0, %1, off" :: "v"(ldsa), "v"(ga) : "memory");
  }
#endif

  // phase 0b: kick off async mask tile DMA (32KB); drained before phase 3.
  {
    const unsigned char* gmask = mask + ((size_t)b * S_ + qbase) * S_;
    #pragma unroll
    for (int j = 0; j < 8; ++j) {
      int o = tid * 16 + j * 4096;
      unsigned ldsa = (unsigned)(uintptr_t)(mk + o);
      unsigned long long ga = (unsigned long long)(uintptr_t)(gmask + o);
      asm volatile("global_load_async_to_lds_b128 %0, %1, off" :: "v"(ldsa), "v"(ga) : "memory");
    }
  }

#if HAVE_TDM
  if (wave == 0) __builtin_amdgcn_s_wait_tensorcnt(0);
#else
  asm volatile("s_wait_asynccnt 0x7" ::: "memory");  // Q-tile b128 (first issued) done
#endif
  __syncthreads();

  // phases 1+2: wave holds fixed mt; A fragments are loop-invariant across N tiles.
  {
    int mt = wave & 1;
    v16bf a0 = load_a_bf(qt, DH_, mt * 16, 0, lane);
    v16bf a1 = load_a_bf(qt, DH_, mt * 16, 32, lane);
    int mhi = (lane >> 4) << 3;

    // phase 1: qb = Qtile(32x64) x bias_table^T(64x1104)
    for (int nt = wave >> 1; nt < NB2P_ / 16; nt += 4) {
      v16bf b0 = load_b_bf(BTbf, DH_, nt * 16, 0, lane);
      v16bf b1 = load_b_bf(BTbf, DH_, nt * 16, 32, lane);
      v8f acc = {};
      acc = wmma_bf16(a0, b0, acc);
      acc = wmma_bf16(a1, b1, acc);
      int n = nt * 16 + (lane & 15);
      #pragma unroll
      for (int r = 0; r < 8; ++r) qb[(mt * 16 + r + mhi) * NB2P_ + n] = acc[r];
    }
    // phase 2: scores = Qtile(32x64) x K^T(64x1024)
    for (int nt = wave >> 1; nt < S_ / 16; nt += 4) {
      __builtin_prefetch(Krows + (size_t)((nt + 4) * 16 + (lane & 15)) * DH_, 0, 1);
      v16bf b0 = load_b_bf(Krows, DH_, nt * 16, 0, lane);
      v16bf b1 = load_b_bf(Krows, DH_, nt * 16, 32, lane);
      v8f acc = {};
      acc = wmma_bf16(a0, b0, acc);
      acc = wmma_bf16(a1, b1, acc);
      int n = nt * 16 + (lane & 15);
      #pragma unroll
      for (int r = 0; r < 8; ++r) sc[(mt * 16 + r + mhi) * S_ + n] = acc[r];
    }
  }
  asm volatile("s_wait_asynccnt 0x0" ::: "memory");  // mask tile resident
  __syncthreads();

  // phase 3: add bucketed bias (gather from qb), scale, apply mask (from LDS)
  const float NEG = -3.402823466e38f;
  const float scale = 0.044194173824159216f;   // DIM^-0.5
  for (int i = tid; i < QT_ * S_; i += 256) {
    int q = i >> 10, k = i & (S_ - 1);
    int gq = qbase + q;
    int q0 = gq / qs1, q1 = gq - q0 * qs1;
    int kk0 = k / ks1, kk1 = k - kk0 * ks1;
    int rv = kk0 - q0, rh = kk1 - q1;
    int av = rv < 0 ? -rv : rv, ah = rh < 0 ? -rh : rh;
    int bucket = (av + ah <= NB_ / 2) ? (rv + NB_ / 2) * NB_ + (rh + NB_ / 2) : 0;
    float v = (sc[i] + qb[q * NB2P_ + bucket]) * scale;
    sc[i] = mk[i] ? v : NEG;
  }
  __syncthreads();

  // phase 4: per-row exact top-k threshold (iterative max extraction, wave32),
  //          then masked softmax numerators into sc, row sums into LDS.
  int kk = topk; if (kk > S_) kk = S_; if (kk < 1) kk = 1;
  for (int rr = 0; rr < QT_ / 8; ++rr) {       // 4 rows per wave
    int row = wave * (QT_ / 8) + rr;
    float vals[32];
    #pragma unroll
    for (int i = 0; i < 32; ++i) vals[i] = sc[row * S_ + lane + 32 * i];
    unsigned removed = 0;
    float rowmax = NEG, thres = NEG;
    for (int t = 0; t < kk; ++t) {
      float lm = NEG; int li = -1;
      #pragma unroll
      for (int i = 0; i < 32; ++i) {
        bool live = !((removed >> i) & 1u);
        if (live && vals[i] > lm) { lm = vals[i]; li = i; }
      }
      float gm = lm;
      #pragma unroll
      for (int off = 16; off > 0; off >>= 1) gm = fmaxf(gm, __shfl_xor(gm, off, 32));
      unsigned long long bal = __ballot((lm == gm) && (li >= 0));
      int src = (int)__ffsll(bal) - 1;
      if (lane == src) removed |= (1u << li);
      if (t == 0) rowmax = gm;
      thres = gm;
    }
    float sum = 0.f;
    #pragma unroll
    for (int i = 0; i < 32; ++i) {
      float v = vals[i];
      float p = (v >= thres) ? __expf(v - rowmax) : 0.f;
      sum += p;
      sc[row * S_ + lane + 32 * i] = p;
    }
    #pragma unroll
    for (int off = 16; off > 0; off >>= 1) sum += __shfl_xor(sum, off, 32);
    if (lane == 0) rowsum[row] = sum;
  }
  __syncthreads();

  // phase 5: out = P(32x1024) x V(1024x64) [WMMA], pipelined B fragments,
  // normalize by rowsum, store bf16.
  for (int t = wave; t < 8; t += 8) {          // 2 m-tiles x 4 n-tiles, one per wave
    int mt = t & 1, nt = t >> 1;
    v8f acc = {};
    v16bf bv = load_b_bf(Vrows, S_, nt * 16, 0, lane);
    for (int k0 = 0; k0 < S_; k0 += 32) {
      int kn = (k0 + 32 < S_) ? (k0 + 32) : k0;
      v16bf bn = load_b_bf(Vrows, S_, nt * 16, kn, lane);
      v16bf a  = load_a_f32(sc, S_, mt * 16, k0, lane);
      acc = wmma_bf16(a, bv, acc);
      bv = bn;
    }
    int d = nt * 16 + (lane & 15), mhi = (lane >> 4) << 3;
    #pragma unroll
    for (int r = 0; r < 8; ++r) {
      int m = mt * 16 + r + mhi;
      int gq = qbase + m;
      float o = acc[r] / rowsum[m];
      AObf[((size_t)b * S_ + gq) * (H_ * DH_) + (h << 6) + d] = f2bf(o);
    }
  }
}

// ---------------------------------------------------------------------------
// K3: output projection (8192 x 512 x 512). Each wave: 16x64 strip, pipelined.
// out[b,s,d] = sum_e AO[b,s,e]*Wo[d,e]  (Wo row d is contiguous in e).
// ---------------------------------------------------------------------------
__global__ void k_out(const unsigned short* __restrict__ AObf, const unsigned short* __restrict__ Wobf,
                      float* __restrict__ out) {
  int wave = blockIdx.x * (blockDim.x >> 5) + (threadIdx.x >> 5);
  int lane = threadIdx.x & 31;
  const int tilesN = DIM_ / 64;  // 8 strips
  int mt = wave / tilesN, nt = wave - mt * tilesN;
  int m0 = mt * 16, n0 = nt * 64;
  v8f acc0 = {}, acc1 = {}, acc2 = {}, acc3 = {};
  v16bf a  = load_a_bf(AObf, DIM_, m0, 0, lane);
  v16bf b0 = load_b_bf(Wobf, DIM_, n0,      0, lane);
  v16bf b1 = load_b_bf(Wobf, DIM_, n0 + 16, 0, lane);
  v16bf b2 = load_b_bf(Wobf, DIM_, n0 + 32, 0, lane);
  v16bf b3 = load_b_bf(Wobf, DIM_, n0 + 48, 0, lane);
  for (int k0 = 0; k0 < DIM_; k0 += 32) {
    int kn = (k0 + 32 < DIM_) ? (k0 + 32) : k0;
    v16bf an  = load_a_bf(AObf, DIM_, m0, kn, lane);
    v16bf bn0 = load_b_bf(Wobf, DIM_, n0,      kn, lane);
    v16bf bn1 = load_b_bf(Wobf, DIM_, n0 + 16, kn, lane);
    v16bf bn2 = load_b_bf(Wobf, DIM_, n0 + 32, kn, lane);
    v16bf bn3 = load_b_bf(Wobf, DIM_, n0 + 48, kn, lane);
    acc0 = wmma_bf16(a, b0, acc0);
    acc1 = wmma_bf16(a, b1, acc1);
    acc2 = wmma_bf16(a, b2, acc2);
    acc3 = wmma_bf16(a, b3, acc3);
    a = an; b0 = bn0; b1 = bn1; b2 = bn2; b3 = bn3;
  }
  int mhi = (lane >> 4) << 3;
  #pragma unroll
  for (int r = 0; r < 8; ++r) {
    int m = m0 + r + mhi;
    float* o = out + (size_t)m * DIM_ + (lane & 15);
    o[n0]      = acc0[r];
    o[n0 + 16] = acc1[r];
    o[n0 + 32] = acc2[r];
    o[n0 + 48] = acc3[r];
  }
}

// ---------------------------------------------------------------------------
// Launch
// ---------------------------------------------------------------------------
extern "C" void kernel_launch(void* const* d_in, const int* in_sizes, int n_in,
                              void* d_out, int out_size, void* d_ws, size_t ws_size,
                              hipStream_t stream) {
  (void)in_sizes; (void)n_in; (void)out_size; (void)ws_size;
  const float* hs   = (const float*)d_in[0];
  const float* wqkv = (const float*)d_in[1];
  const float* wo   = (const float*)d_in[2];
  const float* bt   = (const float*)d_in[3];
  const unsigned char* mask = (const unsigned char*)d_in[4];
  const int* qs1p  = (const int*)d_in[6];
  const int* ks1p  = (const int*)d_in[8];
  const int* topkp = (const int*)d_in[9];

  char* ws = (char*)d_ws;
  size_t off = 0;
  auto carve = [&](size_t bytes) -> char* {
    char* p = ws + off;
    off += (bytes + 255) & ~(size_t)255;
    return p;
  };
  // ~44.5 MB of workspace total
  unsigned short* Hbf  = (unsigned short*)carve((size_t)B_ * S_ * DIM_ * 2);
  unsigned short* Wqbf = (unsigned short*)carve((size_t)E_ * DIM_ * 2);
  unsigned short* Wobf = (unsigned short*)carve((size_t)DIM_ * DIM_ * 2);
  unsigned short* BTbf = (unsigned short*)carve((size_t)NB2P_ * DH_ * 2);
  unsigned short* Qbf  = (unsigned short*)carve((size_t)B_ * H_ * S_ * DH_ * 2);
  unsigned short* Kbf  = (unsigned short*)carve((size_t)B_ * H_ * S_ * DH_ * 2);
  unsigned short* Vt   = (unsigned short*)carve((size_t)B_ * H_ * S_ * DH_ * 2);
  unsigned short* AObf = (unsigned short*)carve((size_t)B_ * S_ * H_ * DH_ * 2);

  k_convert<<<2048, 256, 0, stream>>>(hs, wqkv, wo, bt, Hbf, Wqbf, Wobf, BTbf);

  k_qkv<<<(B_ * S_ / 16) * (E_ / 64) / 8, 256, 0, stream>>>(Hbf, Wqbf, Qbf, Kbf, Vt);

  size_t smem = (size_t)QT_ * S_ * 4      // sc
              + (size_t)QT_ * NB2P_ * 4   // qb
              + (size_t)QT_ * DH_ * 2     // qt
              + (size_t)QT_ * 4           // rowsum
              + (size_t)QT_ * S_;         // mask tile  == 309,376 B (< 320KB/WGP)
  k_attn<<<B_ * H_ * (S_ / QT_), 256, smem, stream>>>(Qbf, Kbf, Vt, BTbf, mask, AObf,
                                                      qs1p, ks1p, topkp);

  k_out<<<(B_ * S_ / 16) * (DIM_ / 64) / 8, 256, 0, stream>>>(AObf, Wobf, (float*)d_out);
}
